// RWKVMixer_71597104825031
// MI455X (gfx1250) — matrix operand order repaired and checked by reading, not compiled, not run on hardware
//
#include <hip/hip_runtime.h>
#include <cstdint>
#include <cstddef>

// ---------------- problem constants ----------------
#define DECAY 0.95f
#define OMD   0.05f       // 1 - DECAY
#define LNEPS 1e-5f

constexpr int Bc = 4;
constexpr int Tc = 4096;
constexpr int Dc = 1024;
constexpr int Lc = 2;
constexpr int Mc = Bc * Tc;            // 16384 rows

constexpr int BLK_M = 128;
constexpr int BLK_N = 64;
constexpr int BLK_K = 64;              // staged K depth (2 WMMA k-substeps)
constexpr int LDS_PAD = 8;             // bf16 elems of row padding
constexpr int LDA = BLK_K + LDS_PAD;   // 72 elems = 144B row stride (16B aligned)

typedef __bf16 bf16;
typedef __attribute__((ext_vector_type(4)))  __bf16 bf16x4;
typedef __attribute__((ext_vector_type(8)))  __bf16 bf16x8;
typedef __attribute__((ext_vector_type(16))) __bf16 v16bf;
typedef __attribute__((ext_vector_type(8)))  float  v8f;

// ---------------- CDNA5 async copy: global -> LDS, tracked by ASYNCcnt ----------
__device__ __forceinline__ uint32_t lds_addr32(const void* p) {
    return (uint32_t)(uintptr_t)p;     // low 32 bits of flat LDS pointer = LDS offset
}
__device__ __forceinline__ void async_copy_b128(uint32_t lds, uint64_t gaddr) {
    asm volatile("global_load_async_to_lds_b128 %0, %1, off"
                 :: "v"(lds), "v"(gaddr) : "memory");
}
__device__ __forceinline__ void wait_async0() {
    asm volatile("s_wait_asynccnt 0" ::: "memory");
}

// ---------------- weight f32 -> bf16 ----------------
__global__ void rwkv_cvt_bf16(const float* __restrict__ src, bf16* __restrict__ dst, int n) {
    int i = (blockIdx.x * blockDim.x + threadIdx.x) * 4;
    if (i < n) {
        float4 v = *(const float4*)(src + i);
        bf16x4 o = { (bf16)v.x, (bf16)v.y, (bf16)v.z, (bf16)v.w };
        *(bf16x4*)(dst + i) = o;
    }
}

// ---------------- fused LayerNorm -> bf16 ----------------
__global__ void rwkv_ln_bf16(const float* __restrict__ x,
                             const float* __restrict__ g,
                             const float* __restrict__ bta,
                             bf16* __restrict__ out) {
    __shared__ float s1[256];
    __shared__ float s2[256];
    const int row = blockIdx.x;
    const int tid = threadIdx.x;
    float4 v = ((const float4*)(x + (size_t)row * Dc))[tid];
    s1[tid] = v.x + v.y + v.z + v.w;
    s2[tid] = v.x * v.x + v.y * v.y + v.z * v.z + v.w * v.w;
    __syncthreads();
    for (int o = 128; o > 0; o >>= 1) {
        if (tid < o) { s1[tid] += s1[tid + o]; s2[tid] += s2[tid + o]; }
        __syncthreads();
    }
    const float mu   = s1[0] * (1.0f / Dc);
    const float var  = s2[0] * (1.0f / Dc) - mu * mu;
    const float rstd = rsqrtf(var + LNEPS);
    float4 gw = ((const float4*)g)[tid];
    float4 bw = ((const float4*)bta)[tid];
    float n0 = (v.x - mu) * rstd * gw.x + bw.x;
    float n1 = (v.y - mu) * rstd * gw.y + bw.y;
    float n2 = (v.z - mu) * rstd * gw.z + bw.z;
    float n3 = (v.w - mu) * rstd * gw.w + bw.w;
    bf16x4 o = { (bf16)n0, (bf16)n1, (bf16)n2, (bf16)n3 };
    ((bf16x4*)(out + (size_t)row * Dc))[tid] = o;
}

// helper: assemble v16bf from two 16B LDS chunks
__device__ __forceinline__ v16bf frag16(const bf16* p0, const bf16* p1) {
    bf16x8 lo = *(const bf16x8*)p0;
    bf16x8 hi = *(const bf16x8*)p1;
    return __builtin_shufflevector(lo, hi, 0,1,2,3,4,5,6,7,8,9,10,11,12,13,14,15);
}

// ---------------- fused K/V/R GEMM (async double-buffered, K=64 tiles) ----------
// C[w] = A(normed bf16) x W[w]^T ; epilogue: kv = k*v (f32), r = sigmoid(r) (f32)
__global__ __launch_bounds__(256)
void rwkv_kvr_gemm(const bf16* __restrict__ A,
                   const bf16* __restrict__ Wk,
                   const bf16* __restrict__ Wv,
                   const bf16* __restrict__ Wr,
                   float* __restrict__ kvout,
                   float* __restrict__ rout) {
    __shared__ bf16 lA[2][BLK_M][LDA];       // 2 x 128 x 72
    __shared__ bf16 lB[2][3][BLK_N][LDA];    // 2 x 3 x 64 x 72

    const int tid  = threadIdx.x;
    const int wave = tid >> 5;
    const int lane = tid & 31;
    const int half = lane >> 4;
    const int l16  = lane & 15;
    const int mBase = blockIdx.y * BLK_M;
    const int nBase = blockIdx.x * BLK_N;

    // staging: 16B chunks; 8 chunks per 64-elem row
    // A: 128 rows * 8 = 1024 chunks -> 4 per thread; B: 64*8=512 per weight -> 2 per thread
    uint64_t gA[4]; uint32_t dA[2][4];
    #pragma unroll
    for (int it = 0; it < 4; ++it) {
        const int c = tid + it * 256;
        const int r = c >> 3, col = (c & 7) * 8;
        gA[it] = (uint64_t)(uintptr_t)(A + (size_t)(mBase + r) * Dc + col);
        dA[0][it] = lds_addr32(&lA[0][r][col]);
        dA[1][it] = lds_addr32(&lA[1][r][col]);
    }
    const bf16* Wp[3] = { Wk, Wv, Wr };
    uint64_t gW[3][2]; uint32_t dW[2][3][2];
    #pragma unroll
    for (int w = 0; w < 3; ++w)
        #pragma unroll
        for (int it = 0; it < 2; ++it) {
            const int c = tid + it * 256;
            const int r = c >> 3, col = (c & 7) * 8;
            gW[w][it] = (uint64_t)(uintptr_t)(Wp[w] + (size_t)(nBase + r) * Dc + col);
            dW[0][w][it] = lds_addr32(&lB[0][w][r][col]);
            dW[1][w][it] = lds_addr32(&lB[1][w][r][col]);
        }

    v8f acc[3][4];
    #pragma unroll
    for (int w = 0; w < 3; ++w)
        #pragma unroll
        for (int nt = 0; nt < 4; ++nt)
            #pragma unroll
            for (int i = 0; i < 8; ++i) acc[w][nt][i] = 0.0f;

    auto stage = [&](int buf) {
        #pragma unroll
        for (int it = 0; it < 4; ++it) {
            async_copy_b128(dA[buf][it], gA[it]);
            gA[it] += BLK_K * sizeof(bf16);
        }
        #pragma unroll
        for (int w = 0; w < 3; ++w)
            #pragma unroll
            for (int it = 0; it < 2; ++it) {
                async_copy_b128(dW[buf][w][it], gW[w][it]);
                gW[w][it] += BLK_K * sizeof(bf16);
            }
        __builtin_prefetch((const void*)(uintptr_t)(gA[0] + BLK_K), 0, 3);
        __builtin_prefetch((const void*)(uintptr_t)(gW[0][0] + BLK_K), 0, 3);
    };

    stage(0);
    wait_async0();
    __syncthreads();

    int cur = 0;
    for (int k0 = 0; k0 < Dc; k0 += BLK_K) {
        if (k0 + BLK_K < Dc) stage(cur ^ 1);     // DMA next tile while computing

        const int mloc = wave * 16 + l16;
        #pragma unroll
        for (int ks = 0; ks < BLK_K; ks += 32) {
            v16bf afrag = frag16(&lA[cur][mloc][ks + half * 8],
                                 &lA[cur][mloc][ks + 16 + half * 8]);
            #pragma unroll
            for (int w = 0; w < 3; ++w) {
                v16bf bf[4];
                #pragma unroll
                for (int nt = 0; nt < 4; ++nt) {
                    const int nloc = nt * 16 + l16;
                    bf[nt] = frag16(&lB[cur][w][nloc][ks + half * 8],
                                    &lB[cur][w][nloc][ks + 16 + half * 8]);
                }
                #pragma unroll
                for (int nt = 0; nt < 4; ++nt)
                    acc[w][nt] = __builtin_amdgcn_wmma_f32_16x16x32_bf16(
                        false, afrag, false, bf[nt], (short)0, acc[w][nt], false, false);
            }
        }
        wait_async0();       // own async copies into next buffer landed
        __syncthreads();     // everyone's landed; cur buffer free for reuse
        cur ^= 1;
    }

    #pragma unroll
    for (int nt = 0; nt < 4; ++nt) {
        const int col = nBase + nt * 16 + l16;
        #pragma unroll
        for (int j = 0; j < 8; ++j) {
            const int row = mBase + wave * 16 + half * 8 + j;
            const size_t off = (size_t)row * Dc + col;
            float kk = acc[0][nt][j];
            float vv = acc[1][nt][j];
            float rr = acc[2][nt][j];
            kvout[off] = kk * vv;
            rout[off]  = 1.0f / (1.0f + __expf(-rr));
        }
    }
}

// ---------------- sequential EMA scan + gate; writes y as bf16 ----------------
__global__ void rwkv_scan(const float* __restrict__ kv,
                          const float* __restrict__ r,
                          bf16* __restrict__ y) {
    const int idx = blockIdx.x * blockDim.x + threadIdx.x;   // < B*D
    const int b = idx >> 10;
    const int d = idx & (Dc - 1);
    const float* kp = kv + (size_t)b * Tc * Dc + d;
    const float* rp = r  + (size_t)b * Tc * Dc + d;
    bf16*        yp = y  + (size_t)b * Tc * Dc + d;
    float s = 0.0f;
    for (int t = 0; t < Tc; t += 4) {
        float k0 = kp[(size_t)(t + 0) * Dc];
        float k1 = kp[(size_t)(t + 1) * Dc];
        float k2 = kp[(size_t)(t + 2) * Dc];
        float k3 = kp[(size_t)(t + 3) * Dc];
        float r0 = rp[(size_t)(t + 0) * Dc];
        float r1 = rp[(size_t)(t + 1) * Dc];
        float r2 = rp[(size_t)(t + 2) * Dc];
        float r3 = rp[(size_t)(t + 3) * Dc];
        s = DECAY * s + OMD * k0; yp[(size_t)(t + 0) * Dc] = (bf16)(r0 * s);
        s = DECAY * s + OMD * k1; yp[(size_t)(t + 1) * Dc] = (bf16)(r1 * s);
        s = DECAY * s + OMD * k2; yp[(size_t)(t + 2) * Dc] = (bf16)(r2 * s);
        s = DECAY * s + OMD * k3; yp[(size_t)(t + 3) * Dc] = (bf16)(r3 * s);
    }
}

// ---------------- output GEMM with residual accumulate (async, K=64 tiles) ------
__global__ __launch_bounds__(256)
void rwkv_out_gemm(const bf16* __restrict__ A,
                   const bf16* __restrict__ Wo,
                   float* __restrict__ out) {
    __shared__ bf16 lA[2][BLK_M][LDA];
    __shared__ bf16 lB[2][BLK_N][LDA];

    const int tid  = threadIdx.x;
    const int wave = tid >> 5;
    const int lane = tid & 31;
    const int half = lane >> 4;
    const int l16  = lane & 15;
    const int mBase = blockIdx.y * BLK_M;
    const int nBase = blockIdx.x * BLK_N;

    uint64_t gA[4]; uint32_t dA[2][4];
    #pragma unroll
    for (int it = 0; it < 4; ++it) {
        const int c = tid + it * 256;
        const int r = c >> 3, col = (c & 7) * 8;
        gA[it] = (uint64_t)(uintptr_t)(A + (size_t)(mBase + r) * Dc + col);
        dA[0][it] = lds_addr32(&lA[0][r][col]);
        dA[1][it] = lds_addr32(&lA[1][r][col]);
    }
    uint64_t gW[2]; uint32_t dW[2][2];
    #pragma unroll
    for (int it = 0; it < 2; ++it) {
        const int c = tid + it * 256;
        const int r = c >> 3, col = (c & 7) * 8;
        gW[it] = (uint64_t)(uintptr_t)(Wo + (size_t)(nBase + r) * Dc + col);
        dW[0][it] = lds_addr32(&lB[0][r][col]);
        dW[1][it] = lds_addr32(&lB[1][r][col]);
    }

    v8f acc[4];
    #pragma unroll
    for (int nt = 0; nt < 4; ++nt)
        #pragma unroll
        for (int i = 0; i < 8; ++i) acc[nt][i] = 0.0f;

    auto stage = [&](int buf) {
        #pragma unroll
        for (int it = 0; it < 4; ++it) {
            async_copy_b128(dA[buf][it], gA[it]);
            gA[it] += BLK_K * sizeof(bf16);
        }
        #pragma unroll
        for (int it = 0; it < 2; ++it) {
            async_copy_b128(dW[buf][it], gW[it]);
            gW[it] += BLK_K * sizeof(bf16);
        }
        __builtin_prefetch((const void*)(uintptr_t)(gW[0] + BLK_K), 0, 3);
    };

    stage(0);
    wait_async0();
    __syncthreads();

    int cur = 0;
    for (int k0 = 0; k0 < Dc; k0 += BLK_K) {
        if (k0 + BLK_K < Dc) stage(cur ^ 1);

        const int mloc = wave * 16 + l16;
        #pragma unroll
        for (int ks = 0; ks < BLK_K; ks += 32) {
            v16bf afrag = frag16(&lA[cur][mloc][ks + half * 8],
                                 &lA[cur][mloc][ks + 16 + half * 8]);
            v16bf bf[4];
            #pragma unroll
            for (int nt = 0; nt < 4; ++nt) {
                const int nloc = nt * 16 + l16;
                bf[nt] = frag16(&lB[cur][nloc][ks + half * 8],
                                &lB[cur][nloc][ks + 16 + half * 8]);
            }
            #pragma unroll
            for (int nt = 0; nt < 4; ++nt)
                acc[nt] = __builtin_amdgcn_wmma_f32_16x16x32_bf16(
                    false, afrag, false, bf[nt], (short)0, acc[nt], false, false);
        }
        wait_async0();
        __syncthreads();
        cur ^= 1;
    }

    #pragma unroll
    for (int nt = 0; nt < 4; ++nt) {
        const int col = nBase + nt * 16 + l16;
        #pragma unroll
        for (int j = 0; j < 8; ++j) {
            const int row = mBase + wave * 16 + half * 8 + j;
            const size_t off = (size_t)row * Dc + col;
            out[off] += acc[nt][j];    // residual accumulate (x + y@Wo^T)
        }
    }
}

// ---------------- host orchestration ----------------
extern "C" void kernel_launch(void* const* d_in, const int* in_sizes, int n_in,
                              void* d_out, int out_size, void* d_ws, size_t ws_size,
                              hipStream_t stream) {
    const float* h    = (const float*)d_in[0];
    const float* ln_w = (const float*)d_in[1];
    const float* ln_b = (const float*)d_in[2];
    const float* Wk   = (const float*)d_in[3];
    const float* Wv   = (const float*)d_in[4];
    const float* Wr   = (const float*)d_in[5];
    const float* Wo   = (const float*)d_in[6];
    float* out = (float*)d_out;

    // workspace layout
    char* ws = (char*)d_ws;
    const size_t wElems = (size_t)Lc * Dc * Dc;          // per-weight elems
    bf16* WkB = (bf16*)ws;
    bf16* WvB = WkB + wElems;
    bf16* WrB = WvB + wElems;
    bf16* WoB = WrB + wElems;
    bf16* act = WoB + wElems;                            // normed / y (bf16) [M,D]
    float* kv = (float*)(act + (size_t)Mc * Dc);         // [M,D] f32
    float* rb = kv + (size_t)Mc * Dc;                    // [M,D] f32

    // convert weights once
    {
        const int n = (int)wElems;
        const int blocks = n / (4 * 256);
        rwkv_cvt_bf16<<<blocks, 256, 0, stream>>>(Wk, WkB, n);
        rwkv_cvt_bf16<<<blocks, 256, 0, stream>>>(Wv, WvB, n);
        rwkv_cvt_bf16<<<blocks, 256, 0, stream>>>(Wr, WrB, n);
        rwkv_cvt_bf16<<<blocks, 256, 0, stream>>>(Wo, WoB, n);
    }

    // residual stream lives in d_out
    hipMemcpyAsync(out, h, (size_t)Mc * Dc * sizeof(float),
                   hipMemcpyDeviceToDevice, stream);

    dim3 gemmGrid(Dc / BLK_N, Mc / BLK_M);   // (16, 128)

    for (int l = 0; l < Lc; ++l) {
        const bf16* wk = WkB + (size_t)l * Dc * Dc;
        const bf16* wv = WvB + (size_t)l * Dc * Dc;
        const bf16* wr = WrB + (size_t)l * Dc * Dc;
        const bf16* wo = WoB + (size_t)l * Dc * Dc;

        rwkv_ln_bf16<<<Mc, 256, 0, stream>>>(out, ln_w + (size_t)l * Dc,
                                             ln_b + (size_t)l * Dc, act);
        rwkv_kvr_gemm<<<gemmGrid, 256, 0, stream>>>(act, wk, wv, wr, kv, rb);
        rwkv_scan<<<(Bc * Dc) / 256, 256, 0, stream>>>(kv, rb, act);
        rwkv_out_gemm<<<gemmGrid, 256, 0, stream>>>(act, wo, out);
    }
}
